// PCFI_19413252178656
// MI455X (gfx1250) — matrix operand order repaired and checked by reading, not compiled, not run on hardware
//
#include <hip/hip_runtime.h>
#include <hip/hip_bf16.h>
#include <stdint.h>

typedef __attribute__((ext_vector_type(16))) _Float16 v16h;
typedef __attribute__((ext_vector_type(8)))  float    v8f;

#define FD 64
#define MAXHOPS 6
#define NITER 8
__device__ constexpr float L09  = -0.15200309344504997f; // log2(0.9)
__device__ constexpr float BETA = 0.5f;

// ---------------- zero / init ----------------
__global__ void k_zero(int* p, long long nwords) {
  long long i = (long long)blockIdx.x * blockDim.x + threadIdx.x;
  long long stride = (long long)gridDim.x * blockDim.x;
  for (; i < nwords; i += stride) p[i] = 0;
}

__global__ void k_init(const float* __restrict__ x, const uint8_t* __restrict__ mask,
                       unsigned long long* __restrict__ reach_prev,
                       unsigned long long* __restrict__ reach_next,
                       uint8_t* __restrict__ dist, float* __restrict__ out0, int N) {
  int n = blockIdx.x * blockDim.x + threadIdx.x;
  if (n >= N) return;
  unsigned long long m = 0ull;
  const uint8_t* mp = mask + (size_t)n * FD;
  const float*   xp = x    + (size_t)n * FD;
  uint8_t* dp = dist + (size_t)n * FD;
  float*   op = out0 + (size_t)n * FD;
  for (int f = 0; f < FD; ++f) {
    bool ob = mp[f] != 0;
    if (ob) m |= (1ull << f);
    dp[f] = ob ? (uint8_t)0 : (uint8_t)255;
    op[f] = ob ? xp[f] : 0.f;
  }
  reach_prev[n] = m;
  reach_next[n] = m;
}

// ---------------- bitmask BFS (64 channels per uint64) ----------------
__global__ void k_bfs_edge(const int* __restrict__ row, const int* __restrict__ col,
                           const unsigned long long* __restrict__ prev,
                           unsigned long long* __restrict__ next, int E) {
  int e = blockIdx.x * blockDim.x + threadIdx.x;
  if (e >= E) return;
  int r = row[e], c = col[e];
  unsigned long long m = prev[r];
  if (m & ~prev[c]) atomicOr(next + c, m);
}

__global__ void k_bfs_node(unsigned long long* __restrict__ prev,
                           const unsigned long long* __restrict__ next,
                           uint8_t* __restrict__ dist, int hop, int N) {
  int n = blockIdx.x * blockDim.x + threadIdx.x;
  if (n >= N) return;
  unsigned long long p = prev[n], q = next[n];
  unsigned long long nw = q & ~p;
  while (nw) {
    int f = __ffsll((unsigned long long)nw) - 1;
    dist[(size_t)n * FD + f] = (uint8_t)hop;
    nw &= nw - 1;
  }
  prev[n] = q;
}

__global__ void k_fix(uint8_t* __restrict__ dist, long long total) {
  long long i = (long long)blockIdx.x * blockDim.x + threadIdx.x;
  long long stride = (long long)gridDim.x * blockDim.x;
  for (; i < total; i += stride)
    if (dist[i] == 255) dist[i] = 0;
}

// ---------------- P factor: P[n][f] = 0.9^(dist+1) ----------------
// weight(e)=0.9^(dc-dn+1)=P[c]*0.9^(-dn); the 0.9^(-dn) factor cancels against
// the same factor inside deg, so propagation needs only P (no exp in hot loop).
__global__ void k_pw(const uint8_t* __restrict__ dist, float* __restrict__ P,
                     long long total) {
  long long i = (long long)blockIdx.x * blockDim.x + threadIdx.x;
  long long stride = (long long)gridDim.x * blockDim.x;
  for (; i < total; i += stride)
    P[i] = exp2f(((float)dist[i] + 1.f) * L09);
}

// ---------------- CSR build ----------------
__global__ void k_count(const int* __restrict__ row, int* __restrict__ cnt, int E) {
  int e = blockIdx.x * blockDim.x + threadIdx.x;
  if (e < E) atomicAdd(&cnt[row[e]], 1);
}

__global__ void k_scan(const int* __restrict__ cnt, int* __restrict__ ptr, int n) {
  __shared__ int tmp[1024];
  __shared__ int running;
  if (threadIdx.x == 0) running = 0;
  __syncthreads();
  for (int base = 0; base < n; base += 1024) {
    int i = base + (int)threadIdx.x;
    int v = (i < n) ? cnt[i] : 0;
    tmp[threadIdx.x] = v;
    __syncthreads();
    for (int offd = 1; offd < 1024; offd <<= 1) {
      int t2 = (threadIdx.x >= (unsigned)offd) ? tmp[threadIdx.x - offd] : 0;
      __syncthreads();
      tmp[threadIdx.x] += t2;
      __syncthreads();
    }
    int run = running;
    if (i < n) ptr[i] = run + tmp[threadIdx.x] - v; // exclusive
    int tot = tmp[1023];
    __syncthreads();
    if (threadIdx.x == 0) running = run + tot;
    __syncthreads();
  }
  if (threadIdx.x == 0) ptr[n] = running;
}

__global__ void k_scatter(const int* __restrict__ row, const int* __restrict__ col,
                          const int* __restrict__ ptr, int* __restrict__ fill,
                          int* __restrict__ adj, int E) {
  int e = blockIdx.x * blockDim.x + threadIdx.x;
  if (e >= E) return;
  int r = row[e];
  int pos = ptr[r] + atomicAdd(&fill[r], 1);
  adj[pos] = col[e];
}

// ------- deg_inv[n][f] = 1/sum_{c in adj(n)} P[c][f] (wave/node, 2 ch/lane) -------
__global__ void k_deg(const int* __restrict__ ptr, const int* __restrict__ adj,
                      const float* __restrict__ P, float* __restrict__ deg_inv, int N) {
  int lane = threadIdx.x & 31;
  int n = blockIdx.x * (blockDim.x >> 5) + (threadIdx.x >> 5);
  if (n >= N) return;
  int f0 = 2 * lane;
  float acc0 = 0.f, acc1 = 0.f;
  int b = ptr[n], en = ptr[n + 1];
  for (int e = b; e < en; ++e) {
    int c = adj[e];
    const float* pc = P + (size_t)c * FD + f0;
    acc0 += pc[0];
    acc1 += pc[1];
  }
  float* dv = deg_inv + (size_t)n * FD + f0;
  dv[0] = acc0 > 0.f ? 1.f / acc0 : 0.f;
  dv[1] = acc1 > 0.f ? 1.f / acc1 : 0.f;
}

// ---------------- propagation step: pure FMA gather, no transcendentals ----------
__global__ void k_prop(const int* __restrict__ ptr, const int* __restrict__ adj,
                       const float* __restrict__ P, const float* __restrict__ deg_inv,
                       const uint8_t* __restrict__ mask, const float* __restrict__ x,
                       const float* __restrict__ cur, float* __restrict__ nxt, int N) {
  int lane = threadIdx.x & 31;
  int n = blockIdx.x * (blockDim.x >> 5) + (threadIdx.x >> 5);
  if (n >= N) return;
  int f0 = 2 * lane;
  float acc0 = 0.f, acc1 = 0.f;
  int b = ptr[n], en = ptr[n + 1];
  if (b < en) {
    int c = adj[b];
    for (int e = b; e < en; ++e) {
      int cn = (e + 1 < en) ? adj[e + 1] : c;
      // prefetch next neighbor's rows (global_prefetch_b8)
      __builtin_prefetch(P   + (size_t)cn * FD + f0, 0, 3);
      __builtin_prefetch(cur + (size_t)cn * FD + f0, 0, 3);
      const float* pc = P   + (size_t)c * FD + f0;
      const float* oc = cur + (size_t)c * FD + f0;
      acc0 += pc[0] * oc[0];
      acc1 += pc[1] * oc[1];
      c = cn;
    }
  }
  size_t base = (size_t)n * FD + f0;
  float v0 = deg_inv[base]     * acc0;
  float v1 = deg_inv[base + 1] * acc1;
  if (mask[base])     v0 = x[base];
  if (mask[base + 1]) v1 = x[base + 1];
  nxt[base] = v0;
  nxt[base + 1] = v1;
}

// ---------------- column mean ----------------
__global__ void k_meansum(const float* __restrict__ out, float* __restrict__ musum, int N) {
  int f = threadIdx.x & 63;
  int rpb = blockDim.x >> 6;
  int r = threadIdx.x >> 6;
  float acc = 0.f;
  for (int n = blockIdx.x * rpb + r; n < N; n += gridDim.x * rpb)
    acc += out[(size_t)n * FD + f];
  atomicAdd(&musum[f], acc);
}

__global__ void k_mu(const float* __restrict__ musum, float* __restrict__ mu, int N) {
  int f = threadIdx.x;
  if (f < FD) mu[f] = musum[f] / (float)N;
}

// ---------------- Gram = (out-mu)^T (out-mu) via WMMA ----------------
__global__ void __launch_bounds__(512) k_gram(const float* __restrict__ out,
                                              const float* __restrict__ mu,
                                              float* __restrict__ G, int N) {
  __shared__ _Float16 cs[32][FD + 8];
  __shared__ float mus[FD];
  int t = threadIdx.x;            // 512 threads = 16 waves
  int lane = t & 31, wv = t >> 5;
  if (t < FD) mus[t] = mu[t];
  __syncthreads();
  int fi = (wv >> 2) << 4, fj = (wv & 3) << 4;
  v8f acc = {};
  int nChunks = (N + 31) >> 5;
  for (int ch = blockIdx.x; ch < nChunks; ch += gridDim.x) {
    int n0 = ch << 5;
    {
      int nl = t >> 4;            // 0..31 node in chunk
      int fb = (t & 15) << 2;     // 4 consecutive features
      int n = n0 + nl;
      float v0 = 0.f, v1 = 0.f, v2 = 0.f, v3 = 0.f;
      if (n < N) {
        const float* p = out + (size_t)n * FD + fb;
        v0 = p[0] - mus[fb]; v1 = p[1] - mus[fb + 1];
        v2 = p[2] - mus[fb + 2]; v3 = p[3] - mus[fb + 3];
      }
      cs[nl][fb]     = (_Float16)v0; cs[nl][fb + 1] = (_Float16)v1;
      cs[nl][fb + 2] = (_Float16)v2; cs[nl][fb + 3] = (_Float16)v3;
    }
    __syncthreads();
    v16h a, bb;
#pragma unroll
    for (int h = 0; h < 16; ++h) {       // A: 16x32, M=feature row, K=node
      int vv = h >> 1, sub = h & 1;
      int K = ((vv < 4) ? 0 : 16) + ((lane < 16) ? 0 : 8) + 2 * (vv & 3) + sub;
      a[h] = cs[K][fi + (lane & 15)];
    }
#pragma unroll
    for (int h = 0; h < 16; ++h) {       // B: 32x16, K=node, N=feature col
      int K = ((h < 8) ? 0 : 16) + ((lane < 16) ? 0 : 8) + (h & 7);
      bb[h] = cs[K][fj + (lane & 15)];
    }
    acc = __builtin_amdgcn_wmma_f32_16x16x32_f16(false, a, false, bb, (short)0, acc,
                                                 false, false);
    __syncthreads();
  }
#pragma unroll
  for (int r = 0; r < 8; ++r) {
    int M = r + ((lane < 16) ? 0 : 8);
    int Nc = lane & 15;
    atomicAdd(&G[(fi + M) * FD + (fj + Nc)], acc[r]);
  }
}

// ---------------- corrcoef normalize -> f16, zero diagonal ----------------
__global__ void k_cor(const float* __restrict__ G, _Float16* __restrict__ cor) {
  for (int idx = threadIdx.x; idx < FD * FD; idx += blockDim.x) {
    int i = idx >> 6, j = idx & 63;
    float den = sqrtf(G[i * FD + i] * G[j * FD + j]);
    float c = (den > 0.f) ? G[idx] / den : 0.f;
    if (!(c == c) || fabsf(c) > 3.0e38f) c = 0.f;  // nan_to_num
    if (i == j) c = 0.f;
    cor[idx] = (_Float16)c;
  }
}

// ---------------- a2 = a1 @ cor via WMMA + fused residual output ----------------
__global__ void __launch_bounds__(256) k_final(const float* __restrict__ out,
                                               const uint8_t* __restrict__ dist,
                                               const float* __restrict__ mu,
                                               const _Float16* __restrict__ cor,
                                               float* __restrict__ res, int N) {
  __shared__ _Float16 a1s[32][FD + 8];
  __shared__ _Float16 cors[FD][FD + 8];
  __shared__ float mus[FD];
  int t = threadIdx.x;            // 256 threads = 8 waves
  int lane = t & 31, wv = t >> 5;
  int n0 = blockIdx.x << 5;       // 32 nodes per block
  if (t < FD) mus[t] = mu[t];
  __syncthreads();
  for (int idx = t; idx < FD * FD; idx += 256)
    cors[idx >> 6][idx & 63] = cor[idx];
  {
    int nl = t >> 3;              // 0..31
    int fb = (t & 7) << 3;        // 8 consecutive features
    int n = n0 + nl;
    for (int q = 0; q < 8; ++q) {
      int f = fb + q;
      float v = 0.f;
      if (n < N) {
        size_t o = (size_t)n * FD + f;
        float d = (float)dist[o];
        v = exp2f(d * L09) * (out[o] - mus[f]);
      }
      a1s[nl][f] = (_Float16)v;
    }
  }
  __syncthreads();
  int mt = wv >> 2;               // 0/1 -> which 16-node tile
  int j = (wv & 3) << 4;          // output column tile
  v8f acc = {};
#pragma unroll
  for (int kc = 0; kc < FD; kc += 32) {
    v16h a, bb;
#pragma unroll
    for (int h = 0; h < 16; ++h) {       // A: 16x32, M=node, K=feature
      int vv = h >> 1, sub = h & 1;
      int K = ((vv < 4) ? 0 : 16) + ((lane < 16) ? 0 : 8) + 2 * (vv & 3) + sub;
      a[h] = a1s[(mt << 4) + (lane & 15)][kc + K];
    }
#pragma unroll
    for (int h = 0; h < 16; ++h) {       // B: 32x16 slice of cor
      int K = ((h < 8) ? 0 : 16) + ((lane < 16) ? 0 : 8) + (h & 7);
      bb[h] = cors[kc + K][j + (lane & 15)];
    }
    acc = __builtin_amdgcn_wmma_f32_16x16x32_f16(false, a, false, bb, (short)0, acc,
                                                 false, false);
  }
#pragma unroll
  for (int r = 0; r < 8; ++r) {
    int n = n0 + (mt << 4) + r + ((lane < 16) ? 0 : 8);
    int col = j + (lane & 15);
    if (n < N) {
      size_t o = (size_t)n * FD + col;
      float ad = exp2f((float)dist[o] * L09);
      res[o] = out[o] + BETA * (1.f - ad) * acc[r];
    }
  }
}

// ---------------- host orchestration ----------------
extern "C" void kernel_launch(void* const* d_in, const int* in_sizes, int n_in,
                              void* d_out, int out_size, void* d_ws, size_t ws_size,
                              hipStream_t stream) {
  const float*   x    = (const float*)d_in[0];
  const int*     ei   = (const int*)d_in[1];
  const uint8_t* mask = (const uint8_t*)d_in[2];
  int N = in_sizes[0] / FD;
  int E = in_sizes[1] / 2;
  const int* row = ei;
  const int* col = ei + E;

  uint8_t* ws = (uint8_t*)d_ws;
  size_t off = 0;
  auto carve = [&](size_t bytes) -> void* {
    off = (off + 255) & ~(size_t)255;
    void* p = ws + off;
    off += bytes;
    return p;
  };

  // contiguous zero region: cnt | fill | musum | mu | G
  long long zwords = 2LL * N + 64 + 64 + FD * FD;
  int*   zbase = (int*)carve((size_t)zwords * 4);
  int*   cnt   = zbase;
  int*   fill  = zbase + N;
  float* musum = (float*)(zbase + 2LL * N);
  float* mu    = musum + 64;
  float* G     = mu + 64;

  int* row_ptr = (int*)carve(((size_t)N + 1) * 4);
  unsigned long long* rprev = (unsigned long long*)carve((size_t)N * 8);
  unsigned long long* rnext = (unsigned long long*)carve((size_t)N * 8);
  uint8_t* dist   = (uint8_t*)carve((size_t)N * FD);
  float* Pfac     = (float*)carve((size_t)N * FD * 4);
  float* deg_inv  = (float*)carve((size_t)N * FD * 4);
  float* outA     = (float*)carve((size_t)N * FD * 4);
  float* outB     = (float*)carve((size_t)N * FD * 4);
  int*   adj      = (int*)carve((size_t)E * 4);
  _Float16* cor16 = (_Float16*)carve((size_t)FD * FD * 2);

  int nb256  = (N + 255) / 256;
  int eb256  = (E + 255) / 256;
  int wavesB = 256 / 32;
  int nbWave = (N + wavesB - 1) / wavesB;

  // 0) zero accumulators (required every call; graph replays don't re-poison)
  k_zero<<<256, 256, 0, stream>>>(zbase, zwords);
  // 1) init reach masks, dist, masked features
  k_init<<<nb256, 256, 0, stream>>>(x, mask, rprev, rnext, dist, outA, N);
  // 2) 6 BFS hops via 64-channel bitmasks
  for (int h = 1; h <= MAXHOPS; ++h) {
    k_bfs_edge<<<eb256, 256, 0, stream>>>(row, col, rprev, rnext, E);
    k_bfs_node<<<nb256, 256, 0, stream>>>(rprev, rnext, dist, h, N);
  }
  // 3) unreached (255) -> 0 per reference
  k_fix<<<512, 256, 0, stream>>>(dist, (long long)N * FD);
  // 3b) P = 0.9^(dist+1): all transcendentals hoisted out of the hot loop
  k_pw<<<512, 256, 0, stream>>>(dist, Pfac, (long long)N * FD);
  // 4) CSR by row
  k_count<<<eb256, 256, 0, stream>>>(row, cnt, E);
  k_scan<<<1, 1024, 0, stream>>>(cnt, row_ptr, N);
  k_scatter<<<eb256, 256, 0, stream>>>(row, col, row_ptr, fill, adj, E);
  // 5) deg_inv = 1 / sum P over neighbors
  k_deg<<<nbWave, 256, 0, stream>>>(row_ptr, adj, Pfac, deg_inv, N);
  // 6) 8 propagation iterations with masked reset (FMA-only inner loop)
  const float* cur = outA;
  float* nxt = outB;
  for (int it = 0; it < NITER; ++it) {
    k_prop<<<nbWave, 256, 0, stream>>>(row_ptr, adj, Pfac, deg_inv, mask, x, cur, nxt, N);
    const float* tmp = cur;
    cur = nxt;
    nxt = (float*)tmp;
  }
  // 7) channel means
  k_meansum<<<256, 256, 0, stream>>>(cur, musum, N);
  k_mu<<<1, 64, 0, stream>>>(musum, mu, N);
  // 8) centered Gram via WMMA -> corrcoef
  k_gram<<<120, 512, 0, stream>>>(cur, mu, G, N);
  k_cor<<<1, 256, 0, stream>>>(G, cor16);
  // 9) a2 = a1 @ cor via WMMA, fused residual output
  k_final<<<(N + 31) / 32, 256, 0, stream>>>(cur, dist, mu, cor16, (float*)d_out, N);
}
